// TFM4488_27DF_28SCF_18CMF_44384192037501
// MI455X (gfx1250) — matrix operand order repaired and checked by reading, not compiled, not run on hardware
//
#include <hip/hip_runtime.h>
#include <hip/hip_bf16.h>

// ---------------- problem constants ----------------
#define B_    2
#define DM    64      // D_MODEL
#define DI    128     // D_INNER
#define DS    16      // D_STATE
#define RT    4       // DT_RANK
#define RR    36      // R = DT_RANK + 2*D_STATE
#define RRP   48      // R padded to multiple of 16 for WMMA N
#define Hh    128
#define Ww    128
#define LL    (Hh*Ww)     // 16384
#define MM    (B_*LL)     // 32768
#define NCHK  64          // scan chunks
#define CH    (LL/NCHK)   // 256 steps per chunk

typedef unsigned int  uint;
typedef unsigned short ushort;
typedef __attribute__((ext_vector_type(16))) __bf16 v16bf;
typedef __attribute__((ext_vector_type(8)))  float  v8f;
typedef __attribute__((ext_vector_type(4)))  uint   u32x4;
typedef __attribute__((ext_vector_type(4)))  int    i32x4;
typedef __attribute__((ext_vector_type(8)))  int    i32x8;

struct U4x2 { uint4 a, b; };   // 32 bytes == v16bf

#ifndef __has_builtin
#define __has_builtin(x) 0
#endif
#if __has_builtin(__builtin_amdgcn_tensor_load_to_lds) && \
    __has_builtin(__builtin_amdgcn_s_wait_tensorcnt)
#define HAVE_TDM 1
#else
#define HAVE_TDM 0
#endif

__device__ __forceinline__ ushort f2bfu(float f) {
    union { float f; uint u; } v; v.f = f;
    uint r = v.u + 0x7FFFu + ((v.u >> 16) & 1u);   // round-to-nearest-even
    return (ushort)(r >> 16);
}
__device__ __forceinline__ float gelu_exact(float x) {
    return 0.5f * x * (1.0f + erff(x * 0.70710678118654752f));
}
__device__ __forceinline__ float softplus_f(float x) {
    return (x > 20.f) ? x : log1pf(__expf(x));
}

// ---------------- weight packing: (Nreal,K) f32 -> fragment-ready bf16 B ----------------
// layout: [kt][nt][lane 0..31][16 halves]; lane holds column n=nt*16+lane%16,
// halves j -> k = kt*32 + (lane>=16 ? 16 : 0) + j
__global__ void pack_b_kernel(const float* __restrict__ W, ushort* __restrict__ out,
                              int K, int N, int Nreal) {
    int idx = blockIdx.x * blockDim.x + threadIdx.x;
    int total = K * N;
    if (idx >= total) return;
    int j    = idx & 15;
    int lane = (idx >> 4) & 31;
    int t    = idx >> 9;
    int Nt   = N >> 4;
    int kt   = t / Nt, nt = t % Nt;
    int n    = (nt << 4) + (lane & 15);
    int k    = (kt << 5) + ((lane & 16) ? 16 : 0) + j;
    float v  = (n < Nreal && k < K) ? W[(size_t)n * K + k] : 0.f;
    out[idx] = f2bfu(v);
}

// ---------------- generic WMMA bf16 GEMM: C[M,N] = A[M,K] * B[K,N] ----------------
// B is staged into LDS once per block via the Tensor Data Mover (TDM), then all
// waves read fragments with ds_load_b128. Each wave computes a 16 x (16*NTW)
// strip so one A fragment feeds NTW back-to-back WMMAs.
// mode bit0: also write bf16 copy to Cbf
// mode bit1: SiLU epilogue
// mode bit2: transposed (B,N,L) store with residual add: out = resid + val
template<int NTW>
__global__ void wmma_gemm(const ushort* __restrict__ A, const ushort* __restrict__ Bp,
                          float* __restrict__ C, ushort* __restrict__ Cbf,
                          const float* __restrict__ resid, float* __restrict__ outT,
                          int M, int K, int N, int mode) {
    __shared__ uint4 sB[2048];                 // 32 KB: max packed-B (K*N bf16)
    int ndw = (K * N) >> 1;                    // packed-B size in dwords
#if HAVE_TDM
    if (threadIdx.x < 32) {
        // ---- Tensor DMA descriptor (D#), 1-D copy of ndw dwords into LDS ----
        unsigned long long ga = (unsigned long long)Bp;
        uint lds = (uint)(size_t)(&sB[0]);     // low 32 bits = LDS byte offset
        u32x4 g0;
        g0[0] = 1u;                            // count=1, is_restore=0, gather off
        g0[1] = lds;                           // lds_addr
        g0[2] = (uint)ga;                      // global_addr[31:0]
        g0[3] = (uint)(ga >> 32) | (2u << 30); // global_addr[56:32] | type=2
        i32x8 g1;
        g1[0] = (int)(2u << 16);               // workgroup_mask=0, data_size=2 (4B)
        g1[1] = (int)((ndw & 0xFFFF) << 16);   // tensor_dim0 low 16
        g1[2] = (int)(((uint)ndw >> 16) | (1u << 16)); // tensor_dim0 hi | tensor_dim1=1
        g1[3] = (int)((uint)(ndw & 0xFFFF) << 16);     // tile_dim0 = ndw
        g1[4] = 1;                             // tile_dim1=1, tile_dim2=0
        g1[5] = ndw;                           // tensor_dim0_stride low
        g1[6] = 0;
        g1[7] = 0;
        i32x4 z4 = {0, 0, 0, 0};
#if __clang_major__ >= 23
        i32x8 z8 = {0, 0, 0, 0, 0, 0, 0, 0};
        __builtin_amdgcn_tensor_load_to_lds(g0, g1, z4, z4, z8, 0);
#else
        __builtin_amdgcn_tensor_load_to_lds(g0, g1, z4, z4, 0);
#endif
        __builtin_amdgcn_s_wait_tensorcnt(0);
    }
    __syncthreads();
#else
    for (int i = threadIdx.x; i < (ndw >> 2); i += blockDim.x)
        sB[i] = ((const uint4*)Bp)[i];
    __syncthreads();
#endif

    int wave = blockIdx.x * (blockDim.x >> 5) + (threadIdx.x >> 5);
    int lane = threadIdx.x & 31;
    int Mt = M >> 4, Nt = N >> 4;
    int Ntg = Nt / NTW;
    if (wave >= Mt * Ntg) return;
    int mt  = wave / Ntg;
    int nt0 = (wave % Ntg) * NTW;
    int row = (mt << 4) + (lane & 15);
    int kb  = (lane & 16) ? 16 : 0;    // A layout: lower lanes K 0-15, upper K 16-31
    v8f acc[NTW];
#pragma unroll
    for (int j = 0; j < NTW; ++j) acc[j] = (v8f){};
    int Kt = K >> 5;
    for (int kt = 0; kt < Kt; ++kt) {
        const ushort* ap = A + (size_t)row * K + (kt << 5) + kb;
        U4x2 ar;
        ar.a = *(const uint4*)(ap);
        ar.b = *(const uint4*)(ap + 8);
        v16bf af = __builtin_bit_cast(v16bf, ar);
#pragma unroll
        for (int j = 0; j < NTW; ++j) {
            const uint4* bp = sB + (((kt * Nt + nt0 + j) * 32 + lane) << 1);
            U4x2 br; br.a = bp[0]; br.b = bp[1];
            v16bf bf = __builtin_bit_cast(v16bf, br);
            acc[j] = __builtin_amdgcn_wmma_f32_16x16x32_bf16(false, af, false, bf,
                                                             (short)0, acc[j], false, false);
        }
    }
    int mrow0 = (mt << 4) + ((lane & 16) >> 1);   // +8 for upper half-wave
#pragma unroll
    for (int j = 0; j < NTW; ++j) {
        int ncol = ((nt0 + j) << 4) + (lane & 15);
#pragma unroll
        for (int v = 0; v < 8; ++v) {
            int m = mrow0 + v;
            float val = acc[j][v];
            if (mode & 2) val = val / (1.f + __expf(-val));        // SiLU
            if (mode & 4) {                                        // (B,N,L) + residual
                int b = m >> 14, l = m & (LL - 1);
                size_t o = ((size_t)b * N + ncol) * LL + l;
                outT[o] = resid[o] + val;
            } else {
                size_t o = (size_t)m * N + ncol;
                C[o] = val;
                if (mode & 1) Cbf[o] = f2bfu(val);
            }
        }
    }
}

// ---------------- channel LayerNorm (over C=64) -> bf16 (M,64), optional FiLM mod ----
__global__ void ln_channel_kernel(const float* __restrict__ x, const float* __restrict__ w,
                                  const float* __restrict__ bb, const float* __restrict__ ci,
                                  ushort* __restrict__ actA, int withMod) {
    int m = blockIdx.x * blockDim.x + threadIdx.x;
    if (m >= MM) return;
    int b = m >> 14, l = m & (LL - 1);
    const float* xp = x + (size_t)b * DM * LL + l;
    float s = 0.f, s2 = 0.f;
    for (int c = 0; c < DM; ++c) { float v = xp[(size_t)c * LL]; s += v; s2 += v * v; }
    float mu  = s * (1.f / DM);
    float var = s2 * (1.f / DM) - mu * mu;
    float inv = rsqrtf(var + 1e-5f);
    ushort* op = actA + (size_t)m * DM;
    for (int c = 0; c < DM; ++c) {
        float v = (xp[(size_t)c * LL] - mu) * inv * w[c] + bb[c];
        if (withMod) {
            float sc = ci[(size_t)m * 128 + c];
            float sh = ci[(size_t)m * 128 + 64 + c];
            v = v * (1.f + sc) + sh;
        }
        op[c] = f2bfu(v);
    }
}

// ---------------- depthwise 3x3 conv + bias + GELU; emits f32 + bf16 (M,128) ---------
__global__ void dwconv2d_gelu(const float* __restrict__ xz, const float* __restrict__ cw,
                              const float* __restrict__ cb, float* __restrict__ xsf,
                              ushort* __restrict__ xsb) {
    int idx = blockIdx.x * blockDim.x + threadIdx.x;     // M*128
    if (idx >= MM * DI) return;
    int c = idx & 127, m = idx >> 7;
    int b = m >> 14, l = m & (LL - 1);
    int h = l >> 7, wp = l & 127;
    float acc = cb[c];
#pragma unroll
    for (int dy = -1; dy <= 1; ++dy) {
        int hh = h + dy; if ((unsigned)hh >= Hh) continue;
#pragma unroll
        for (int dx = -1; dx <= 1; ++dx) {
            int ww2 = wp + dx; if ((unsigned)ww2 >= Ww) continue;
            acc += cw[c * 9 + (dy + 1) * 3 + (dx + 1)]
                 * xz[(((size_t)b * LL + hh * Ww + ww2) * 256) + c];
        }
    }
    float g = gelu_exact(acc);
    xsf[idx] = g;
    xsb[idx] = f2bfu(g);
}

// ---------------- dwconv1d(k=7, SAME over L) + dt proj + softplus -------------------
__global__ void xconv_dt(const float* __restrict__ xdbl, const float* __restrict__ xcw,
                         const float* __restrict__ xcb, const float* __restrict__ dtw,
                         const float* __restrict__ dtb, float* __restrict__ delta,
                         float* __restrict__ Bsv, float* __restrict__ Csv) {
    int m = blockIdx.x * blockDim.x + threadIdx.x;
    if (m >= MM) return;
    int b = m >> 14, l = m & (LL - 1);
    float tmp[RR];
    for (int r = 0; r < RR; ++r) {
        float a = xcb[r];
#pragma unroll
        for (int t = 0; t < 7; ++t) {
            int ll = l + t - 3;
            if ((unsigned)ll < LL) a += xcw[r * 7 + t] * xdbl[((size_t)b * LL + ll) * RRP + r];
        }
        tmp[r] = a;
    }
#pragma unroll
    for (int n = 0; n < DS; ++n) {
        Bsv[(size_t)m * DS + n] = tmp[RT + n];
        Csv[(size_t)m * DS + n] = tmp[RT + DS + n];
    }
    for (int d = 0; d < DI; ++d) {
        float s = dtb[d];
#pragma unroll
        for (int r = 0; r < RT; ++r) s += dtw[d * RT + r] * tmp[r];
        delta[(size_t)m * DI + d] = softplus_f(s);
    }
}

// ---------------- chunked selective scan ------------------------------------------
// pass A: per (b,chunk,d) compute segment aggregate (prod a, acc b) per state n
__global__ void scanA(const float* __restrict__ delta, const float* __restrict__ u,
                      const float* __restrict__ Bsv, const float* __restrict__ Alogs,
                      float* __restrict__ apd, float* __restrict__ bcd) {
    __shared__ float sB[CH * DS];                  // 16 KB
    int b = blockIdx.x / NCHK, chn = blockIdx.x % NCHK;
    int d = threadIdx.x;
    int l0 = chn * CH;
    for (int i = threadIdx.x; i < CH * DS; i += blockDim.x)
        sB[i] = Bsv[((size_t)b * LL + l0) * DS + i];
    __syncthreads();
    float An[DS], ap[DS], bc[DS];
#pragma unroll
    for (int n = 0; n < DS; ++n) { An[n] = -__expf(Alogs[d * DS + n]); ap[n] = 1.f; bc[n] = 0.f; }
    for (int i = 0; i < CH; ++i) {
        size_t off = ((size_t)b * LL + l0 + i) * DI + d;
        float dv = delta[off];
        float du = dv * u[off];
        const float* bl = sB + i * DS;
#pragma unroll
        for (int n = 0; n < DS; ++n) {
            float a = __expf(dv * An[n]);
            bc[n] = a * bc[n] + du * bl[n];
            ap[n] *= a;
        }
    }
    size_t o = (((size_t)b * NCHK + chn) * DI + d) * DS;
#pragma unroll
    for (int n = 0; n < DS; ++n) { apd[o + n] = ap[n]; bcd[o + n] = bc[n]; }
}

// pass B: exclusive scan over chunks per (b,d,n) -> initial state h0 per chunk
__global__ void scanB(const float* __restrict__ apd, const float* __restrict__ bcd,
                      float* __restrict__ h0) {
    int t = blockIdx.x * blockDim.x + threadIdx.x;
    if (t >= B_ * DI * DS) return;
    int b = t / (DI * DS), rem = t % (DI * DS);
    float h = 0.f;
    for (int c = 0; c < NCHK; ++c) {
        size_t idx = ((size_t)b * NCHK + c) * (DI * DS) + rem;
        h0[idx] = h;
        h = apd[idx] * h + bcd[idx];
    }
}

// pass C: replay each chunk with correct h0, emit y = C·h + D*u
__global__ void scanC(const float* __restrict__ delta, const float* __restrict__ u,
                      const float* __restrict__ Bsv, const float* __restrict__ Csv,
                      const float* __restrict__ Alogs, const float* __restrict__ Dsv,
                      const float* __restrict__ h0, float* __restrict__ y) {
    __shared__ float sB[CH * DS];
    __shared__ float sC[CH * DS];
    int b = blockIdx.x / NCHK, chn = blockIdx.x % NCHK;
    int d = threadIdx.x;
    int l0 = chn * CH;
    for (int i = threadIdx.x; i < CH * DS; i += blockDim.x) {
        size_t g = ((size_t)b * LL + l0) * DS + i;
        sB[i] = Bsv[g];
        sC[i] = Csv[g];
    }
    __syncthreads();
    float An[DS], h[DS];
    size_t ho = (((size_t)b * NCHK + chn) * DI + d) * DS;
#pragma unroll
    for (int n = 0; n < DS; ++n) { An[n] = -__expf(Alogs[d * DS + n]); h[n] = h0[ho + n]; }
    float Dd = Dsv[d];
    for (int i = 0; i < CH; ++i) {
        size_t off = ((size_t)b * LL + l0 + i) * DI + d;
        float dv = delta[off];
        float uv = u[off];
        float du = dv * uv;
        const float* bl = sB + i * DS;
        const float* cl = sC + i * DS;
        float yv = 0.f;
#pragma unroll
        for (int n = 0; n < DS; ++n) {
            float a = __expf(dv * An[n]);
            h[n] = a * h[n] + du * bl[n];
            yv += h[n] * cl[n];
        }
        y[off] = yv + uv * Dd;
    }
}

// ---------------- LayerNorm over D_INNER + GELU(z) gate -> bf16 (M,128) ------------
__global__ void ln_gate(const float* __restrict__ y, const float* __restrict__ xz,
                        const float* __restrict__ w, const float* __restrict__ bb,
                        ushort* __restrict__ ynb) {
    int m = blockIdx.x * blockDim.x + threadIdx.x;
    if (m >= MM) return;
    const float* yp = y + (size_t)m * DI;
    float s = 0.f, s2 = 0.f;
    for (int c = 0; c < DI; ++c) { float v = yp[c]; s += v; s2 += v * v; }
    float mu  = s * (1.f / DI);
    float var = s2 * (1.f / DI) - mu * mu;
    float inv = rsqrtf(var + 1e-5f);
    for (int c = 0; c < DI; ++c) {
        float v  = (yp[c] - mu) * inv * w[c] + bb[c];
        float zz = xz[(size_t)m * 256 + 128 + c];
        ynb[(size_t)m * DI + c] = f2bfu(v * gelu_exact(zz));
    }
}

// =====================================================================================
extern "C" void kernel_launch(void* const* d_in, const int* in_sizes, int n_in,
                              void* d_out, int out_size, void* d_ws, size_t ws_size,
                              hipStream_t stream) {
    (void)in_sizes; (void)n_in; (void)out_size; (void)ws_size;

    // -------- input unpacking (setup_inputs insertion order, nested dicts flattened) --
    const float* x_img = (const float*)d_in[0];
    const float* x_evt = (const float*)d_in[1];
    const float* nimw  = (const float*)d_in[2];
    const float* nimb  = (const float*)d_in[3];
    const float* nevw  = (const float*)d_in[4];
    const float* nevb  = (const float*)d_in[5];
    // ssm leaf order: in_proj_w, conv2d_w, conv2d_b, x_proj_weight, x_conv_w, x_conv_b,
    //                 dt_projs_weight, dt_projs_bias, A_logs, Ds, out_norm_w, out_norm_b, out_proj_w
    const float* const* evp = (const float* const*)&d_in[6];     // 13 leaves
    const float* cross_w    = (const float*)d_in[19];
    const float* const* imp = (const float* const*)&d_in[20];    // 13 leaves
    const float* projout_w  = (const float*)d_in[33];

    // -------- workspace arena ---------------------------------------------------------
    size_t off = 0;
    auto alloc = [&](size_t bytes) { size_t o = off; off += (bytes + 255) & ~(size_t)255; return o; };
    char* ws = (char*)d_ws;

    size_t o_pk_inE = alloc(64 * 256 * 2);
    size_t o_pk_xpE = alloc(128 * RRP * 2);
    size_t o_pk_opE = alloc(128 * 64 * 2);
    size_t o_pk_inI = alloc(64 * 256 * 2);
    size_t o_pk_xpI = alloc(128 * RRP * 2);
    size_t o_pk_opI = alloc(128 * 64 * 2);
    size_t o_pk_cr  = alloc(64 * 128 * 2);
    size_t o_pk_po  = alloc(64 * 64 * 2);

    size_t o_actA = alloc((size_t)MM * DM * 2);
    size_t o_xz   = alloc((size_t)MM * 256 * 4);
    size_t o_xsf  = alloc((size_t)MM * DI * 4);
    size_t o_xsb  = alloc((size_t)MM * DI * 2);
    size_t o_xdbl = alloc((size_t)MM * RRP * 4);
    size_t o_dlt  = alloc((size_t)MM * DI * 4);
    size_t o_Bs   = alloc((size_t)MM * DS * 4);
    size_t o_Cs   = alloc((size_t)MM * DS * 4);
    size_t o_apd  = alloc((size_t)B_ * NCHK * DI * DS * 4);
    size_t o_bcd  = alloc((size_t)B_ * NCHK * DI * DS * 4);
    size_t o_h0   = alloc((size_t)B_ * NCHK * DI * DS * 4);
    size_t o_y    = alloc((size_t)MM * DI * 4);   // reused as `ci` after event SS2D
    size_t o_ynb  = alloc((size_t)MM * DI * 2);
    size_t o_outf = alloc((size_t)MM * DM * 4);
    size_t o_evb  = alloc((size_t)MM * DM * 2);
    size_t o_imb  = alloc((size_t)MM * DM * 2);

    const int TB = 256;
    auto pack = [&](const float* W, size_t o, int K, int N, int Nreal) {
        int total = K * N;
        pack_b_kernel<<<(total + TB - 1) / TB, TB, 0, stream>>>(W, (ushort*)(ws + o), K, N, Nreal);
    };
    pack(evp[0],   o_pk_inE, 64, 256, 256);
    pack(evp[3],   o_pk_xpE, 128, RRP, RR);
    pack(evp[12],  o_pk_opE, 128, 64, 64);
    pack(imp[0],   o_pk_inI, 64, 256, 256);
    pack(imp[3],   o_pk_xpI, 128, RRP, RR);
    pack(imp[12],  o_pk_opI, 128, 64, 64);
    pack(cross_w,  o_pk_cr,  64, 128, 128);
    pack(projout_w,o_pk_po,  64, 64, 64);

    auto gemm = [&](size_t oA, size_t oB, float* C, ushort* Cbf, const float* resid,
                    float* outT, int M, int K, int N, int mode) {
        int Nt = N >> 4;
        if ((Nt & 3) == 0) {
            int waves = (M >> 4) * (Nt >> 2);
            wmma_gemm<4><<<waves / 8, 256, 0, stream>>>((const ushort*)(ws + oA),
                                                        (const ushort*)(ws + oB),
                                                        C, Cbf, resid, outT, M, K, N, mode);
        } else {   // Nt == 3 (x_proj, N = 48)
            int waves = (M >> 4) * (Nt / 3);
            wmma_gemm<3><<<waves / 8, 256, 0, stream>>>((const ushort*)(ws + oA),
                                                        (const ushort*)(ws + oB),
                                                        C, Cbf, resid, outT, M, K, N, mode);
        }
    };

    auto run_ss2d = [&](const float* const* p, size_t pk_in, size_t pk_xp, size_t pk_op,
                        ushort* out_bf) {
        float*  xz   = (float*)(ws + o_xz);
        float*  xsf  = (float*)(ws + o_xsf);
        float*  xdbl = (float*)(ws + o_xdbl);
        float*  dlt  = (float*)(ws + o_dlt);
        float*  Bs   = (float*)(ws + o_Bs);
        float*  Cs   = (float*)(ws + o_Cs);
        float*  y    = (float*)(ws + o_y);

        // in_proj GEMM: (M,64) x (64,256)
        gemm(o_actA, pk_in, xz, nullptr, nullptr, nullptr, MM, 64, 256, 0);
        // dwconv2d 3x3 + GELU
        dwconv2d_gelu<<<(MM * DI) / TB, TB, 0, stream>>>(xz, p[1], p[2], xsf,
                                                         (ushort*)(ws + o_xsb));
        // x_proj GEMM: (M,128) x (128,48)
        gemm(o_xsb, pk_xp, xdbl, nullptr, nullptr, nullptr, MM, 128, RRP, 0);
        // dwconv1d k=7 + dt projection + softplus
        xconv_dt<<<MM / TB, TB, 0, stream>>>(xdbl, p[4], p[5], p[6], p[7], dlt, Bs, Cs);
        // chunked selective scan
        scanA<<<B_ * NCHK, DI, 0, stream>>>(dlt, xsf, Bs, p[8],
                                            (float*)(ws + o_apd), (float*)(ws + o_bcd));
        scanB<<<(B_ * DI * DS) / TB, TB, 0, stream>>>((float*)(ws + o_apd),
                                                      (float*)(ws + o_bcd),
                                                      (float*)(ws + o_h0));
        scanC<<<B_ * NCHK, DI, 0, stream>>>(dlt, xsf, Bs, Cs, p[8], p[9],
                                            (float*)(ws + o_h0), y);
        // out_norm + GELU(z) gate -> bf16
        ln_gate<<<MM / TB, TB, 0, stream>>>(y, xz, p[10], p[11], (ushort*)(ws + o_ynb));
        // out_proj GEMM: (M,128) x (128,64), keep bf16 copy
        gemm(o_ynb, pk_op, (float*)(ws + o_outf), out_bf, nullptr, nullptr, MM, 128, 64, 1);
    };

    // ---- event path ----
    ln_channel_kernel<<<MM / TB, TB, 0, stream>>>(x_evt, nevw, nevb, nullptr,
                                                  (ushort*)(ws + o_actA), 0);
    run_ss2d(evp, o_pk_inE, o_pk_xpE, o_pk_opE, (ushort*)(ws + o_evb));

    // ---- cross interaction: ci = SiLU(event_out @ cross_w^T), stored in y-buffer ----
    gemm(o_evb, o_pk_cr, (float*)(ws + o_y), nullptr, nullptr, nullptr, MM, 64, 128, 2);

    // ---- img path: LN + FiLM modulation with (scale, shift) from ci ----
    ln_channel_kernel<<<MM / TB, TB, 0, stream>>>(x_img, nimw, nimb, (const float*)(ws + o_y),
                                                  (ushort*)(ws + o_actA), 1);
    run_ss2d(imp, o_pk_inI, o_pk_xpI, o_pk_opI, (ushort*)(ws + o_imb));

    // ---- final: out = x_img + img_out @ proj_out_w^T, stored back as (B,C,H,W) ----
    gemm(o_imb, o_pk_po, nullptr, nullptr, x_img, (float*)d_out, MM, 64, 64, 4);
}
